// DecoderLayerWithMOE_39376260170307
// MI455X (gfx1250) — compile-verified, matrix operand order checked by reading, not run on hardware
//
#include <hip/hip_runtime.h>

typedef __bf16 bf16;
typedef __attribute__((ext_vector_type(16))) __bf16 v16bf;
typedef __attribute__((ext_vector_type(8)))  __bf16 v8bf;
typedef __attribute__((ext_vector_type(8)))  float  v8f;
typedef __attribute__((ext_vector_type(4))) unsigned int u32x4;
typedef __attribute__((ext_vector_type(8))) int i32x8;
typedef __attribute__((ext_vector_type(4))) int i32x4;

#define DEV __device__ __forceinline__

// Tensor Data Mover availability (device pass only; fallback path otherwise).
#if defined(__HIP_DEVICE_COMPILE__) && defined(__gfx1250__) && \
    __has_builtin(__builtin_amdgcn_tensor_load_to_lds) &&      \
    __has_builtin(__builtin_amdgcn_s_wait_tensorcnt)
#define USE_TDM 1
#else
#define USE_TDM 0
#endif

// DPP cross-lane reductions (16-lane butterfly with xor {15,7,3,1}).
#if defined(__HIP_DEVICE_COMPILE__) && defined(__gfx1250__) && \
    __has_builtin(__builtin_amdgcn_mov_dpp) &&                 \
    __has_builtin(__builtin_amdgcn_mov_dpp8)
#define HAVE_DPP 1
#else
#define HAVE_DPP 0
#endif

static constexpr int S_LEN  = 2048;
static constexpr int BATCH  = 4;
static constexpr int DMODEL = 1024;
static constexpr int NHEAD  = 16;
static constexpr int HDIM   = 64;
static constexpr int NEXP   = 8;
static constexpr int DFFN   = 4096;
static constexpr int MTOK   = S_LEN * BATCH;   // 8192 tokens
static constexpr int NQKV   = 3 * DMODEL;      // 3072

// ---------------------------------------------------------------------------
// WMMA helpers (CDNA5 16x16x32 bf16, fp32 accumulate)
// A-frag layout (ISA 7.12.2): lane m=(lane&15) holds row m; K-chunks
// {8*(lane>>4)..+7} and {16+8*(lane>>4)..+7}. B-frag mirrors with N per lane.
// ---------------------------------------------------------------------------
DEV v16bf pack16(const bf16* p0, const bf16* p1) {
  v8bf lo = *(const v8bf*)p0;
  v8bf hi = *(const v8bf*)p1;
  v16bf f;
#pragma unroll
  for (int i = 0; i < 8; ++i) { f[i] = lo[i]; f[i + 8] = hi[i]; }
  return f;
}

DEV v16bf load_frag(const bf16* tile, int ldr, int lane) {
  const bf16* p = tile + (size_t)(lane & 15) * ldr + ((lane >> 4) << 3);
  return pack16(p, p + 16);
}

DEV v8f wmma_bf16(v16bf a, v16bf b, v8f c) {
  return __builtin_amdgcn_wmma_f32_16x16x32_bf16(false, a, false, b,
                                                 (short)0, c, false, false);
}

// ---------------------------------------------------------------------------
// 16-lane-group reductions (groups = lanes 0-15 / 16-31).
// DPP path: ROW_MIRROR(0x140)=xor15, ROW_HALF_MIRROR(0x141)=xor7, DPP8 xor3/1.
// ---------------------------------------------------------------------------
constexpr int dpp8sel(int a, int b, int c, int d, int e, int f, int g, int h) {
  return a | (b << 3) | (c << 6) | (d << 9) | (e << 12) | (f << 15) |
         (g << 18) | (h << 21);
}
constexpr int DPP8_XOR3 = dpp8sel(3, 2, 1, 0, 7, 6, 5, 4);
constexpr int DPP8_XOR1 = dpp8sel(1, 0, 3, 2, 5, 4, 7, 6);

DEV float red16_max(float v) {
#if HAVE_DPP
  v = fmaxf(v, __int_as_float(__builtin_amdgcn_mov_dpp(__float_as_int(v), 0x140, 0xf, 0xf, true)));
  v = fmaxf(v, __int_as_float(__builtin_amdgcn_mov_dpp(__float_as_int(v), 0x141, 0xf, 0xf, true)));
  v = fmaxf(v, __int_as_float(__builtin_amdgcn_mov_dpp8(__float_as_int(v), DPP8_XOR3)));
  v = fmaxf(v, __int_as_float(__builtin_amdgcn_mov_dpp8(__float_as_int(v), DPP8_XOR1)));
#else
#pragma unroll
  for (int m = 1; m < 16; m <<= 1) v = fmaxf(v, __shfl_xor(v, m, 32));
#endif
  return v;
}

DEV float red16_sum(float v) {
#if HAVE_DPP
  v += __int_as_float(__builtin_amdgcn_mov_dpp(__float_as_int(v), 0x140, 0xf, 0xf, true));
  v += __int_as_float(__builtin_amdgcn_mov_dpp(__float_as_int(v), 0x141, 0xf, 0xf, true));
  v += __int_as_float(__builtin_amdgcn_mov_dpp8(__float_as_int(v), DPP8_XOR3));
  v += __int_as_float(__builtin_amdgcn_mov_dpp8(__float_as_int(v), DPP8_XOR1));
#else
#pragma unroll
  for (int m = 1; m < 16; m <<= 1) v += __shfl_xor(v, m, 32);
#endif
  return v;
}

#if USE_TDM
// ---------------------------------------------------------------------------
// TDM 2D tile load Global->LDS  (CDNA5 ISA).
// data_size=2B.  wElems x hRows tile; tensor row stride = strideElems.
// ---------------------------------------------------------------------------
DEV void tdm_load_2d(unsigned lds_off, const void* gsrc,
                     unsigned wElems, unsigned hRows, unsigned strideElems,
                     unsigned padIntCode, unsigned padAmtCode)
{
  unsigned long long ga = (unsigned long long)(size_t)gsrc;
  u32x4 g0;
  g0.x = 1u;                                   // count=1 valid descriptor
  g0.y = lds_off;                              // lds_addr
  g0.z = (unsigned)ga;                         // global_addr[31:0]
  g0.w = (unsigned)((ga >> 32) & 0x01ffffffu)  // global_addr[56:32]
         | (2u << 30);                         // type=2 ("image")
  i32x8 g1;
  g1[0] = (int)((1u << 16)                     // data_size = 2 bytes
                | (1u << 20)                   // pad_enable
                | (padIntCode << 22)
                | (padAmtCode << 25));
  g1[1] = (int)(wElems << 16);                 // tensor_dim0[15:0]
  g1[2] = (int)(hRows << 16);                  // tensor_dim1[15:0]
  g1[3] = (int)(wElems << 16);                 // tile_dim0
  g1[4] = (int)hRows;                          // tile_dim1 (tile_dim2=0)
  g1[5] = (int)strideElems;                    // tensor_dim0_stride[31:0]
  g1[6] = 0;
  g1[7] = 0;
  i32x4 gz = {0, 0, 0, 0};
#if __clang_major__ >= 23
  i32x8 gz8 = {};
  __builtin_amdgcn_tensor_load_to_lds(g0, g1, gz, gz, gz8, 0);
#else
  __builtin_amdgcn_tensor_load_to_lds(g0, g1, gz, gz, 0);
#endif
}
#endif  // USE_TDM

// ---------------------------------------------------------------------------
// Shared K-loop: acc += A(BMxK-tile) @ W(BNxK-tile)^T, TDM double-buffered.
// Wave computes IM x JN 16x16 subtiles at (wm, wn).
// ---------------------------------------------------------------------------
template <int BM, int BN, int IM, int JN>
DEV void gemm_k_loop(const bf16* __restrict__ Ablk, const bf16* __restrict__ Wblk,
                     int K, int wm, int wn, int tid, v8f (&acc)[IM][JN])
{
  constexpr int BK = 32, LDT = 40;
  const int lane = tid & 31;
  const int wid  = tid >> 5;

  auto compute_step = [&](const bf16* tA, const bf16* tB) {
    v16bf af[IM], bfr[JN];
#pragma unroll
    for (int i = 0; i < IM; ++i)
      af[i] = load_frag(tA + (wm + i * 16) * LDT, LDT, lane);
#pragma unroll
    for (int j = 0; j < JN; ++j)
      bfr[j] = load_frag(tB + (wn + j * 16) * LDT, LDT, lane);
#pragma unroll
    for (int i = 0; i < IM; ++i)
#pragma unroll
      for (int j = 0; j < JN; ++j)
        acc[i][j] = wmma_bf16(af[i], bfr[j], acc[i][j]);
  };

#if USE_TDM
  __shared__ __align__(16) bf16 sA[2][BM * LDT];
  __shared__ __align__(16) bf16 sB[2][BN * LDT];
  if (wid == 0)
    tdm_load_2d((unsigned)(size_t)&sA[0][0], Ablk, BK, BM, K, 3, 3);
  else if (wid == 1)
    tdm_load_2d((unsigned)(size_t)&sB[0][0], Wblk, BK, BN, K, 3, 3);
  __builtin_amdgcn_s_wait_tensorcnt((short)0);
  __syncthreads();
  int buf = 0;
  for (int k0 = 0; k0 < K; k0 += BK) {
    if (k0 + BK < K) {   // issue next tile while computing this one
      if (wid == 0)
        tdm_load_2d((unsigned)(size_t)&sA[buf ^ 1][0], Ablk + k0 + BK, BK, BM, K, 3, 3);
      else if (wid == 1)
        tdm_load_2d((unsigned)(size_t)&sB[buf ^ 1][0], Wblk + k0 + BK, BK, BN, K, 3, 3);
    }
    compute_step(&sA[buf][0], &sB[buf][0]);
    __builtin_amdgcn_s_wait_tensorcnt((short)0);
    __syncthreads();
    buf ^= 1;
  }
#else
  __shared__ __align__(16) bf16 sA[BM * LDT];
  __shared__ __align__(16) bf16 sB[BN * LDT];
  for (int k0 = 0; k0 < K; k0 += BK) {
    for (int c = tid; c < BM * 2; c += 256) {
      int row = c >> 1, kh = (c & 1) * 16;
      *(v8bf*)&sA[row * LDT + kh]     = *(const v8bf*)(Ablk + (size_t)row * K + k0 + kh);
      *(v8bf*)&sA[row * LDT + kh + 8] = *(const v8bf*)(Ablk + (size_t)row * K + k0 + kh + 8);
    }
    for (int c = tid; c < BN * 2; c += 256) {
      int row = c >> 1, kh = (c & 1) * 16;
      *(v8bf*)&sB[row * LDT + kh]     = *(const v8bf*)(Wblk + (size_t)row * K + k0 + kh);
      *(v8bf*)&sB[row * LDT + kh + 8] = *(const v8bf*)(Wblk + (size_t)row * K + k0 + kh + 8);
    }
    __syncthreads();
    compute_step(sA, sB);
    __syncthreads();
  }
#endif
}

// ---------------------------------------------------------------------------
// Generic WMMA GEMM:  out[M,N] = A[M,K] @ W[N,K]^T + bias[N]
// ---------------------------------------------------------------------------
template <bool RELU, bool STF32, bool STBF16>
__global__ __launch_bounds__(256) void gemm_wmma(
    const bf16* __restrict__ A, const bf16* __restrict__ W,
    const float* __restrict__ bias,
    float* __restrict__ outf, bf16* __restrict__ outb,
    int M, int N, int K)
{
  const int tid  = threadIdx.x;
  const int lane = tid & 31;
  const int wid  = tid >> 5;
  const int wm   = (wid >> 2) * 64;
  const int wn   = (wid & 3) * 32;
  const int bm   = blockIdx.y * 128;
  const int bn   = blockIdx.x * 128;

  v8f acc[4][2] = {};
  gemm_k_loop<128, 128, 4, 2>(A + (size_t)bm * K, W + (size_t)bn * K,
                              K, wm, wn, tid, acc);

  const int rbase = bm + wm + ((lane >> 4) << 3);
  const int cbase = bn + wn + (lane & 15);
#pragma unroll
  for (int i = 0; i < 4; ++i)
#pragma unroll
    for (int j = 0; j < 2; ++j)
#pragma unroll
      for (int r = 0; r < 8; ++r) {
        int row = rbase + i * 16 + r;
        int col = cbase + j * 16;
        float v = acc[i][j][r] + bias[col];
        if (RELU) v = fmaxf(v, 0.0f);
        size_t idx = (size_t)row * N + col;
        if (STF32) outf[idx] = v;
        if (STBF16) outb[idx] = (bf16)v;
      }
}

// ---------------------------------------------------------------------------
// Fused dense soft-MoE: out[M,N] = sum_e gate[row,e]*relu(A @ W_e^T + b_e)
// Block tile 64x128; loops experts with registers holding both the per-expert
// accumulator and the gated sum -> single store, no global RMW traffic.
// ---------------------------------------------------------------------------
__global__ __launch_bounds__(256) void moe_gemm_wmma(
    const bf16* __restrict__ A, const bf16* __restrict__ Wall,
    const float* __restrict__ biasAll, const float* __restrict__ gate,
    float* __restrict__ outf, int M, int N, int K)
{
  const int tid  = threadIdx.x;
  const int lane = tid & 31;
  const int wid  = tid >> 5;
  const int wm   = (wid >> 2) * 32;   // 2 wave-rows x 32
  const int wn   = (wid & 3) * 32;    // 4 wave-cols x 32
  const int bm   = blockIdx.y * 64;
  const int bn   = blockIdx.x * 128;

  const int rbase = bm + wm + ((lane >> 4) << 3);
  const int cbase = bn + wn + (lane & 15);

  v8f fin[2][2] = {};
  for (int e = 0; e < NEXP; ++e) {
    v8f acc[2][2] = {};
    gemm_k_loop<64, 128, 2, 2>(A + (size_t)bm * K,
                               Wall + ((size_t)e * N + bn) * K,
                               K, wm, wn, tid, acc);
    const float* bias = biasAll + (size_t)e * N;
#pragma unroll
    for (int i = 0; i < 2; ++i)
#pragma unroll
      for (int j = 0; j < 2; ++j)
#pragma unroll
        for (int r = 0; r < 8; ++r) {
          int row = rbase + i * 16 + r;
          int col = cbase + j * 16;
          float v = fmaxf(acc[i][j][r] + bias[col], 0.0f);
          fin[i][j][r] += gate[(size_t)row * NEXP + e] * v;
        }
  }
#pragma unroll
  for (int i = 0; i < 2; ++i)
#pragma unroll
    for (int j = 0; j < 2; ++j)
#pragma unroll
      for (int r = 0; r < 8; ++r) {
        int row = rbase + i * 16 + r;
        int col = cbase + j * 16;
        outf[(size_t)row * N + col] = fin[i][j][r];
      }
}

// ---------------------------------------------------------------------------
// Flash attention: block = (128 q-rows, one (b,h)); each wave owns 16 q-rows.
// Online softmax with DPP reductions; V tile staged via TDM overlapped with
// Q@K^T; P staged per-wave in LDS.
// ---------------------------------------------------------------------------
__global__ __launch_bounds__(256) void attn_flash(
    const bf16* __restrict__ qkv, bf16* __restrict__ ctxout)
{
  __shared__ __align__(16) bf16 sV[32 * 72];        // 32 keys x 64 hd (pad 72)
  __shared__ __align__(16) bf16 sPb[8 * 16 * 40];   // per-wave P 16x32 (pad 40)

  const int tid  = threadIdx.x;
  const int lane = tid & 31;
  const int wid  = tid >> 5;
  const int lh   = lane >> 4;
  const int ln   = lane & 15;
  const int bh   = blockIdx.y;
  const int b    = bh / NHEAD;
  const int h    = bh % NHEAD;
  const int q0   = blockIdx.x * 128 + wid * 16;

  v16bf qf0, qf1;
  {
    size_t t = (size_t)(q0 + ln) * BATCH + b;
    const bf16* base = qkv + t * NQKV + h * HDIM + lh * 8;
    qf0 = pack16(base, base + 16);
    qf1 = pack16(base + 32, base + 48);
  }

  float mrow[8], lrow[8];
#pragma unroll
  for (int r = 0; r < 8; ++r) { mrow[r] = -3.0e38f; lrow[r] = 0.0f; }
  v8f ctx[4] = {};

  bf16* pb = &sPb[wid * (16 * 40)];

  for (int kb = 0; kb < S_LEN; kb += 32) {
    __syncthreads();   // prior-iteration readers of sV / sPb are done

#if USE_TDM
    if (wid == 0)
      tdm_load_2d((unsigned)(size_t)sV,
                  qkv + ((size_t)kb * BATCH + b) * NQKV + 2 * DMODEL + h * HDIM,
                  HDIM, 32, (unsigned)(BATCH * NQKV), 4, 3);
#else
    {
      int vr = tid >> 3, vc = (tid & 7) * 8;
      size_t t = (size_t)(kb + vr) * BATCH + b;
      *(v8bf*)&sV[vr * 72 + vc] =
          *(const v8bf*)(qkv + t * NQKV + 2 * DMODEL + h * HDIM + vc);
    }
    __syncthreads();
#endif

    // S tiles: two 16-key subtiles, contract over hd=64 (2 wmma each).
    v8f s0 = {}, s1 = {};
    {
      size_t t0 = (size_t)(kb + ln) * BATCH + b;
      const bf16* kp0 = qkv + t0 * NQKV + DMODEL + h * HDIM + lh * 8;
      s0 = wmma_bf16(qf0, pack16(kp0, kp0 + 16), s0);
      s0 = wmma_bf16(qf1, pack16(kp0 + 32, kp0 + 48), s0);
      size_t t1 = (size_t)(kb + 16 + ln) * BATCH + b;
      const bf16* kp1 = qkv + t1 * NQKV + DMODEL + h * HDIM + lh * 8;
      s1 = wmma_bf16(qf0, pack16(kp1, kp1 + 16), s1);
      s1 = wmma_bf16(qf1, pack16(kp1 + 32, kp1 + 48), s1);
    }

    // online softmax update over the 32 new keys (DPP reductions)
#pragma unroll
    for (int r = 0; r < 8; ++r) {
      float a0 = s0[r] * 0.125f;                 // 1/sqrt(64)
      float a1 = s1[r] * 0.125f;
      float mx = red16_max(fmaxf(a0, a1));
      float mn = fmaxf(mrow[r], mx);
      float al = __expf(mrow[r] - mn);
      float p0 = __expf(a0 - mn);
      float p1 = __expf(a1 - mn);
      float ps = red16_sum(p0 + p1);
      lrow[r] = lrow[r] * al + ps;
      mrow[r] = mn;
#pragma unroll
      for (int j = 0; j < 4; ++j) ctx[j][r] *= al;
      int prow = r + lh * 8;
      pb[prow * 40 + ln]      = (bf16)p0;
      pb[prow * 40 + 16 + ln] = (bf16)p1;
    }

#if USE_TDM
    __builtin_amdgcn_s_wait_tensorcnt((short)0);
#endif
    __syncthreads();   // sV staged + sPb visible

    // ctx += P(16x32) @ V(32x64)
    v16bf pf = load_frag(pb, 40, lane);
#pragma unroll
    for (int j = 0; j < 4; ++j) {
      int n  = j * 16 + ln;
      int ko = lh * 8;
      v16bf vf;
#pragma unroll
      for (int i = 0; i < 8; ++i) {
        vf[i]     = sV[(ko + i) * 72 + n];
        vf[i + 8] = sV[(16 + ko + i) * 72 + n];
      }
      ctx[j] = wmma_bf16(pf, vf, ctx[j]);
    }
  }

#pragma unroll
  for (int j = 0; j < 4; ++j)
#pragma unroll
    for (int r = 0; r < 8; ++r) {
      size_t t = (size_t)(q0 + r + lh * 8) * BATCH + b;
      ctxout[t * DMODEL + h * HDIM + j * 16 + ln] = (bf16)(ctx[j][r] / lrow[r]);
    }
}

// ---------------------------------------------------------------------------
// Residual + LayerNorm (D=1024), one block per row; bf16 side-copy optional.
// ---------------------------------------------------------------------------
__global__ __launch_bounds__(256) void add_ln_kernel(
    const float* __restrict__ A, const float* __restrict__ Bv,
    const float* __restrict__ g, const float* __restrict__ be,
    float* __restrict__ outf, bf16* __restrict__ outb)
{
  const int row = blockIdx.x, tid = threadIdx.x;
  const int lane = tid & 31, wid = tid >> 5;
  const float* pa = A + (size_t)row * DMODEL;
  const float* pb = Bv + (size_t)row * DMODEL;
  float v[4], s = 0.f, ss = 0.f;
#pragma unroll
  for (int i = 0; i < 4; ++i) {
    int idx = tid + i * 256;
    v[i] = pa[idx] + pb[idx];
    s += v[i]; ss += v[i] * v[i];
  }
#pragma unroll
  for (int m = 1; m < 32; m <<= 1) {
    s += __shfl_xor(s, m, 32);
    ss += __shfl_xor(ss, m, 32);
  }
  __shared__ float rs[8], rss[8];
  if (lane == 0) { rs[wid] = s; rss[wid] = ss; }
  __syncthreads();
  s = 0.f; ss = 0.f;
#pragma unroll
  for (int i = 0; i < 8; ++i) { s += rs[i]; ss += rss[i]; }
  float mu  = s * (1.0f / DMODEL);
  float var = ss * (1.0f / DMODEL) - mu * mu;
  float rsg = rsqrtf(var + 1e-5f);
  float* po = outf + (size_t)row * DMODEL;
#pragma unroll
  for (int i = 0; i < 4; ++i) {
    int idx = tid + i * 256;
    float o = (v[i] - mu) * rsg * g[idx] + be[idx];
    po[idx] = o;
    if (outb) outb[(size_t)row * DMODEL + idx] = (bf16)o;
  }
}

// ---------------------------------------------------------------------------
// Gate: softmax(x @ gate_w^T + gate_b) over E=8; wave w computes expert w.
// ---------------------------------------------------------------------------
__global__ __launch_bounds__(256) void gate_kernel(
    const float* __restrict__ x, const float* __restrict__ gw,
    const float* __restrict__ gb, float* __restrict__ gate)
{
  const int row = blockIdx.x, tid = threadIdx.x;
  const int lane = tid & 31, wid = tid >> 5;
  __shared__ float sd[8];
  const float* px = x + (size_t)row * DMODEL;
  const float* pw = gw + (size_t)wid * DMODEL;
  float acc = 0.f;
  for (int i = lane; i < DMODEL; i += 32) acc += px[i] * pw[i];
#pragma unroll
  for (int m = 1; m < 32; m <<= 1) acc += __shfl_xor(acc, m, 32);
  if (lane == 0) sd[wid] = acc + gb[wid];
  __syncthreads();
  if (tid == 0) {
    float mx = sd[0];
    for (int i = 1; i < 8; ++i) mx = fmaxf(mx, sd[i]);
    float e[8], sm = 0.f;
    for (int i = 0; i < 8; ++i) { e[i] = __expf(sd[i] - mx); sm += e[i]; }
    float inv = 1.0f / sm;
    for (int i = 0; i < 8; ++i) gate[(size_t)row * NEXP + i] = e[i] * inv;
  }
}

// ---------------------------------------------------------------------------
__global__ void f32_to_bf16_kernel(const float* __restrict__ in,
                                   bf16* __restrict__ out, size_t n) {
  size_t i = (size_t)blockIdx.x * blockDim.x + threadIdx.x;
  size_t stride = (size_t)gridDim.x * blockDim.x;
  for (; i < n; i += stride) out[i] = (bf16)in[i];
}

// ---------------------------------------------------------------------------
extern "C" void kernel_launch(void* const* d_in, const int* in_sizes, int n_in,
                              void* d_out, int out_size, void* d_ws,
                              size_t ws_size, hipStream_t stream)
{
  (void)in_sizes; (void)n_in; (void)out_size; (void)ws_size;
  const float* x      = (const float*)d_in[0];
  const float* w_in   = (const float*)d_in[1];
  const float* b_in   = (const float*)d_in[2];
  const float* w_out  = (const float*)d_in[3];
  const float* b_out  = (const float*)d_in[4];
  const float* gate_w = (const float*)d_in[5];
  const float* gate_b = (const float*)d_in[6];
  const float* exp_w  = (const float*)d_in[7];
  const float* exp_b  = (const float*)d_in[8];
  const float* ffn_w1 = (const float*)d_in[9];
  const float* ffn_b1 = (const float*)d_in[10];
  const float* ffn_w2 = (const float*)d_in[11];
  const float* ffn_b2 = (const float*)d_in[12];
  const float* g1  = (const float*)d_in[13];
  const float* be1 = (const float*)d_in[14];
  const float* g2  = (const float*)d_in[15];
  const float* be2 = (const float*)d_in[16];
  const float* g3  = (const float*)d_in[17];
  const float* be3 = (const float*)d_in[18];
  float* outp = (float*)d_out;

  char* wsp = (char*)d_ws;
  size_t off = 0;
  auto alloc = [&](size_t bytes) -> char* {
    char* p = wsp + off;
    off += (bytes + 255) & ~(size_t)255;
    return p;
  };

  bf16*  xb      = (bf16*)alloc((size_t)MTOK * DMODEL * 2);
  bf16*  w_in_b  = (bf16*)alloc((size_t)NQKV * DMODEL * 2);
  bf16*  w_out_b = (bf16*)alloc((size_t)DMODEL * DMODEL * 2);
  bf16*  exp_w_b = (bf16*)alloc((size_t)NEXP * DMODEL * DMODEL * 2);
  bf16*  ffn1_b  = (bf16*)alloc((size_t)DFFN * DMODEL * 2);
  bf16*  ffn2_b  = (bf16*)alloc((size_t)DMODEL * DFFN * 2);
  bf16*  qkv_b   = (bf16*)alloc((size_t)MTOK * NQKV * 2);
  bf16*  ctx_b   = (bf16*)alloc((size_t)MTOK * DMODEL * 2);
  float* attnp   = (float*)alloc((size_t)MTOK * DMODEL * 4);
  float* y1f     = (float*)alloc((size_t)MTOK * DMODEL * 4);
  bf16*  y1b     = (bf16*)alloc((size_t)MTOK * DMODEL * 2);
  float* gatep   = (float*)alloc((size_t)MTOK * NEXP * 4);
  float* moe     = (float*)alloc((size_t)MTOK * DMODEL * 4);
  float* y2f     = (float*)alloc((size_t)MTOK * DMODEL * 4);
  bf16*  y2b     = (bf16*)alloc((size_t)MTOK * DMODEL * 2);
  bf16*  hb      = (bf16*)alloc((size_t)MTOK * DFFN * 2);
  float* ffnf    = (float*)alloc((size_t)MTOK * DMODEL * 4);

  const dim3 blk(256);
  const dim3 cvt(2048);

  // fp32 -> bf16 conversions
  f32_to_bf16_kernel<<<cvt, blk, 0, stream>>>(x, xb, (size_t)MTOK * DMODEL);
  f32_to_bf16_kernel<<<cvt, blk, 0, stream>>>(w_in, w_in_b, (size_t)NQKV * DMODEL);
  f32_to_bf16_kernel<<<cvt, blk, 0, stream>>>(w_out, w_out_b, (size_t)DMODEL * DMODEL);
  f32_to_bf16_kernel<<<cvt, blk, 0, stream>>>(exp_w, exp_w_b, (size_t)NEXP * DMODEL * DMODEL);
  f32_to_bf16_kernel<<<cvt, blk, 0, stream>>>(ffn_w1, ffn1_b, (size_t)DFFN * DMODEL);
  f32_to_bf16_kernel<<<cvt, blk, 0, stream>>>(ffn_w2, ffn2_b, (size_t)DMODEL * DFFN);

  // QKV projection (bf16 out only)
  gemm_wmma<false, false, true><<<dim3(NQKV / 128, MTOK / 128), blk, 0, stream>>>(
      xb, w_in_b, b_in, nullptr, qkv_b, MTOK, NQKV, DMODEL);

  // attention
  attn_flash<<<dim3(S_LEN / 128, BATCH * NHEAD), blk, 0, stream>>>(qkv_b, ctx_b);

  // output projection (f32 out)
  gemm_wmma<false, true, false><<<dim3(DMODEL / 128, MTOK / 128), blk, 0, stream>>>(
      ctx_b, w_out_b, b_out, attnp, nullptr, MTOK, DMODEL, DMODEL);

  // LN1: y1 = LN(x + attn_out)
  add_ln_kernel<<<dim3(MTOK), blk, 0, stream>>>(x, attnp, g1, be1, y1f, y1b);

  // gate softmax
  gate_kernel<<<dim3(MTOK), blk, 0, stream>>>(y1f, gate_w, gate_b, gatep);

  // fused dense soft MoE (all 8 experts, single store)
  moe_gemm_wmma<<<dim3(DMODEL / 128, MTOK / 64), blk, 0, stream>>>(
      y1b, exp_w_b, exp_b, gatep, moe, MTOK, DMODEL, DMODEL);

  // LN2
  add_ln_kernel<<<dim3(MTOK), blk, 0, stream>>>(y1f, moe, g2, be2, y2f, y2b);

  // FFN
  gemm_wmma<true, false, true><<<dim3(DFFN / 128, MTOK / 128), blk, 0, stream>>>(
      y2b, ffn1_b, ffn_b1, nullptr, hb, MTOK, DFFN, DMODEL);
  gemm_wmma<false, true, false><<<dim3(DMODEL / 128, MTOK / 128), blk, 0, stream>>>(
      hb, ffn2_b, ffn_b2, ffnf, nullptr, MTOK, DMODEL, DFFN);

  // LN3 -> final output
  add_ln_kernel<<<dim3(MTOK), blk, 0, stream>>>(y2f, ffnf, g3, be3, outp, nullptr);
}
